// EventAugmentedLSTMCell_75247827026353
// MI455X (gfx1250) — compile-verified, matrix operand order checked by reading, not run on hardware
//
#include <hip/hip_runtime.h>
#include <math.h>

// Problem sizes (fixed by the reference)
#define B_    2048
#define D_    256
#define H_    512
#define N_    64
#define G_    2048   // 4*H
#define FEAT_ 513

// Use the Tensor Data Mover for weight-tile staging (flip to 0 if builtin mismatches)
#ifndef CDNA5_USE_TDM
#define CDNA5_USE_TDM 1
#endif

typedef __attribute__((ext_vector_type(16))) __bf16 bf16x16;
typedef __attribute__((ext_vector_type(8)))  float  f32x8;
typedef __attribute__((ext_vector_type(4)))  unsigned int u32x4;
typedef __attribute__((ext_vector_type(8)))  int i32x8;
typedef __attribute__((ext_vector_type(4)))  int i32x4;

// ---------------- helpers ----------------
__device__ __forceinline__ unsigned short f2bf(float f) {
  return __builtin_bit_cast(unsigned short, (__bf16)f);   // hw v_cvt, RNE
}
__device__ __forceinline__ unsigned pack2bf(float lo, float hi) {
  return (unsigned)f2bf(lo) | ((unsigned)f2bf(hi) << 16);
}
__device__ __forceinline__ float sigf(float x) { return 1.0f / (1.0f + __expf(-x)); }

// A-matrix gathered from up to 3 fp32 row segments (all widths multiples of 32)
struct Seg { const float* p; unsigned long stride; int kBegin; int kEnd; };

// ---------------- generic WMMA GEMM:  out = A @ W^T (+bias) (+dt[m]*wlast[n]) ----------------
// A: M x K fp32 gathered from segments, bf16-converted into LDS (VALU cvt)
// W: Nout x K bf16 row-major; B tile DMA'd into LDS by the Tensor Data Mover
// Block tile 128x128, 256 threads = 8 waves; wave tile 32x64 = 2x4 WMMA 16x16x32 bf16.
// Single-barrier double-buffered K pipeline (frag prefetch to regs, then stage next).
__global__ __launch_bounds__(256)
void wmma_gemm_nt(Seg s0, Seg s1, Seg s2,
                  const unsigned short* __restrict__ Wb,
                  float* __restrict__ out, int ldOut,
                  const float* __restrict__ bias,
                  const float* __restrict__ wlast,
                  const float* __restrict__ dt, long dtStride,
                  int K, int Nout) {
  __shared__ unsigned short As[2][128][32];
  __shared__ unsigned short Bs[2][128][32];

  const int tid  = threadIdx.x;
  const int lane = tid & 31;
  const int wave = tid >> 5;
  const int wm = wave >> 1;            // 0..3 -> 32-row strip
  const int wn = wave & 1;             // 0..1 -> 64-col strip
  const bool w0 = (wave == 0);
  const int blockN = blockIdx.x * 128;
  const int blockM = blockIdx.y * 128;

  f32x8 acc[2][4] = {};

  const int r    = tid >> 1;           // 0..127 (tile row for staging)
  const int half = tid & 1;            // which 16-wide K half this thread stages

  auto stageA = [&](int buf, int kc) {
    const int kk = kc + half * 16;
    const Seg* s = (kk < s0.kEnd) ? &s0 : (kk < s1.kEnd) ? &s1 : &s2;
    const float* src = s->p + (size_t)(blockM + r) * s->stride + (kk - s->kBegin);
    const float4 f0 = ((const float4*)src)[0];
    const float4 f1 = ((const float4*)src)[1];
    const float4 f2 = ((const float4*)src)[2];
    const float4 f3 = ((const float4*)src)[3];
    uint4 pa, pb;
    pa.x = pack2bf(f0.x, f0.y); pa.y = pack2bf(f0.z, f0.w);
    pa.z = pack2bf(f1.x, f1.y); pa.w = pack2bf(f1.z, f1.w);
    pb.x = pack2bf(f2.x, f2.y); pb.y = pack2bf(f2.z, f2.w);
    pb.z = pack2bf(f3.x, f3.y); pb.w = pack2bf(f3.z, f3.w);
    uint4* dst = (uint4*)&As[buf][r][half * 16];
    dst[0] = pa; dst[1] = pb;
  };

#if CDNA5_USE_TDM
  // Issue a TDM 2D tile load: tile 32(K) x 128(rows) of the bf16 weight matrix
  // (tensor: Nout rows x K cols, row stride K) straight into Bs[buf].
  auto tdmB = [&](int buf, int kc) {
    const unsigned long long gaddr =
        (unsigned long long)(const void*)(Wb + (size_t)blockN * K + kc);
    const unsigned lds = (unsigned)(unsigned long long)(const void*)&Bs[buf][0][0];
    u32x4 g0;
    g0.x = 1u;                                    // count=1, user descriptor
    g0.y = lds;                                   // lds_addr
    g0.z = (unsigned)gaddr;                       // global_addr[31:0]
    g0.w = (unsigned)((gaddr >> 32) & 0x01FFFFFFull) | 0x80000000u; // addr[56:32] | type=2
    i32x8 g1;
    g1[0] = (int)(1u << 16);                      // data_size=1 (2 bytes/elem)
    g1[1] = (int)(((unsigned)K & 0xFFFFu) << 16); // tensor_dim0[15:0]
    g1[2] = (int)((((unsigned)K >> 16) & 0xFFFFu) | (((unsigned)Nout & 0xFFFFu) << 16));
    g1[3] = (int)((((unsigned)Nout >> 16) & 0xFFFFu) | (32u << 16)); // tile_dim0=32
    g1[4] = (int)128u;                            // tile_dim1=128 rows, tile_dim2=0
    g1[5] = (int)(unsigned)K;                     // tensor_dim0_stride[31:0]
    g1[6] = 0;                                    // stride hi, dim1_stride lo
    g1[7] = 0;
    const i32x4 z4 = {0, 0, 0, 0};
    const i32x8 z8 = {0, 0, 0, 0, 0, 0, 0, 0};
    __builtin_amdgcn_tensor_load_to_lds(g0, g1, z4, z4, z8, 0);
  };
#else
  auto stageB = [&](int buf, int kc) {
    const unsigned short* src = Wb + (size_t)(blockN + r) * K + kc + half * 16;
    uint4* dst = (uint4*)&Bs[buf][r][half * 16];
    dst[0] = ((const uint4*)src)[0];
    dst[1] = ((const uint4*)src)[1];
    if (kc + 32 < K) __builtin_prefetch(src + 32, 0, 3);  // global_prefetch_b8
  };
#endif

  // ---- prologue: fill buffer 0 ----
  stageA(0, 0);
#if CDNA5_USE_TDM
  if (w0) { tdmB(0, 0); __builtin_amdgcn_s_wait_tensorcnt(0); }
#else
  stageB(0, 0);
#endif
  __syncthreads();

  const int khalf = lane >> 4;   // lanes 0-15: K runs {0..7,16..23}; 16-31: {8..15,24..31}
  const int lrow  = lane & 15;
  int buf = 0;

  for (int kc = 0; kc < K; kc += 32) {
    // ---- pull fragments for this chunk into registers ----
    union AB { bf16x16 v; uint4 q[2]; };
    AB a[2], b[4];
#pragma unroll
    for (int mt = 0; mt < 2; ++mt) {
      const uint4* Ar = (const uint4*)&As[buf][wm * 32 + mt * 16 + lrow][0];
      a[mt].q[0] = Ar[khalf];
      a[mt].q[1] = Ar[2 + khalf];
    }
#pragma unroll
    for (int nt = 0; nt < 4; ++nt) {
      const uint4* Br = (const uint4*)&Bs[buf][wn * 64 + nt * 16 + lrow][0];
      b[nt].q[0] = Br[khalf];
      b[nt].q[1] = Br[2 + khalf];
    }

    // ---- stage next chunk into the other buffer (overlaps with WMMA) ----
    const int nxt = kc + 32;
    if (nxt < K) {
      stageA(buf ^ 1, nxt);
#if CDNA5_USE_TDM
      if (w0) tdmB(buf ^ 1, nxt);
#else
      stageB(buf ^ 1, nxt);
#endif
    }

    // ---- 8 WMMAs on the current fragments ----
#pragma unroll
    for (int mt = 0; mt < 2; ++mt)
#pragma unroll
      for (int nt = 0; nt < 4; ++nt)
        acc[mt][nt] = __builtin_amdgcn_wmma_f32_16x16x32_bf16(
            false, a[mt].v, false, b[nt].v, (short)0, acc[mt][nt], false, false);

#if CDNA5_USE_TDM
    if (w0) __builtin_amdgcn_s_wait_tensorcnt(0);
#endif
    __syncthreads();
    buf ^= 1;
  }

  // ---- epilogue ----
#pragma unroll
  for (int mt = 0; mt < 2; ++mt) {
#pragma unroll
    for (int nt = 0; nt < 4; ++nt) {
      const int gn = blockN + wn * 64 + nt * 16 + lrow;
      const float badd = bias  ? bias[gn]  : 0.0f;
      const float wl   = wlast ? wlast[gn] : 0.0f;
#pragma unroll
      for (int rr = 0; rr < 8; ++rr) {
        const int gm = blockM + wm * 32 + mt * 16 + rr + (khalf ? 8 : 0);
        float v = acc[mt][nt][rr] + badd;
        if (wlast) v += dt[(size_t)gm * dtStride] * wl;   // rank-1 delta_t column
        out[(size_t)gm * ldOut + gn] = v;
      }
    }
  }
}

// ---------------- P = Wk^T @ Wq  (D x D, bf16 output) ----------------
__global__ __launch_bounds__(256)
void compute_P_bf16(const float* __restrict__ Wk, const float* __restrict__ Wq,
                    unsigned short* __restrict__ Pb) {
  __shared__ float ks[16][16];
  __shared__ float qs[16][16];
  const int tx = threadIdx.x & 15, ty = threadIdx.x >> 4;
  const int i0 = blockIdx.y * 16, j0 = blockIdx.x * 16;
  float acc = 0.f;
  for (int h0 = 0; h0 < H_; h0 += 16) {
    ks[ty][tx] = Wk[(size_t)(h0 + ty) * D_ + i0 + tx];
    qs[ty][tx] = Wq[(size_t)(h0 + ty) * D_ + j0 + tx];
    __syncthreads();
#pragma unroll
    for (int hh = 0; hh < 16; ++hh) acc += ks[hh][ty] * qs[hh][tx];
    __syncthreads();
  }
  Pb[(size_t)(i0 + ty) * D_ + j0 + tx] = f2bf(acc);
}

// ---------------- weight concat + fp32->bf16 ----------------
__global__ void cat2_bf16(unsigned short* __restrict__ dst,
                          const float* __restrict__ A, int ldA, int wA,
                          const float* __restrict__ Bsrc, int ldB, int wB, int rows) {
  const int W = wA + wB;
  const size_t total = (size_t)rows * W;
  for (size_t i = (size_t)blockIdx.x * blockDim.x + threadIdx.x; i < total;
       i += (size_t)gridDim.x * blockDim.x) {
    const int row = (int)(i / W), col = (int)(i % W);
    const float v = (col < wA) ? A[(size_t)row * ldA + col]
                               : Bsrc[(size_t)row * ldB + (col - wA)];
    dst[i] = f2bf(v);
  }
}

__global__ void prep_misc(const float* __restrict__ Wih, const float* __restrict__ bih,
                          const float* __restrict__ bhh,
                          float* __restrict__ wlast, float* __restrict__ bias1) {
  const int j = blockIdx.x * blockDim.x + threadIdx.x;
  if (j < G_) {
    wlast[j] = Wih[(size_t)j * FEAT_ + 2 * D_];  // last column (delta_t)
    bias1[j] = bih[j] + bhh[j];
  }
}

// ---------------- slot selection (sims, first-empty / argmax) ----------------
__global__ __launch_bounds__(64)
void select_slots(const float* __restrict__ qk, const float* __restrict__ slots,
                  const float* __restrict__ delta_in, const unsigned char* __restrict__ filled_in,
                  float* __restrict__ dt_out, float* __restrict__ filled_out,
                  int* __restrict__ idxArr) {
  __shared__ float qs[D_];
  __shared__ float sims[N_];
  __shared__ int   emp[N_];
  __shared__ int   sIdx;
  const int b = blockIdx.x;
  const int n = threadIdx.x;
  for (int d = n; d < D_; d += 64) qs[d] = qk[(size_t)b * D_ + d];
  __syncthreads();
  const float* srow = slots + ((size_t)b * N_ + n) * D_;
  float acc = 0.f;
#pragma unroll 4
  for (int d = 0; d < D_; d += 4) {
    const float4 s4 = *(const float4*)(srow + d);
    acc += s4.x * qs[d] + s4.y * qs[d + 1] + s4.z * qs[d + 2] + s4.w * qs[d + 3];
  }
  sims[n] = acc;
  emp[n]  = filled_in[(size_t)b * N_ + n] ? 0 : 1;
  __syncthreads();
  if (n == 0) {
    int idx = -1;
    for (int k = 0; k < N_; ++k) if (emp[k]) { idx = k; break; }
    if (idx < 0) {
      idx = 0; float best = sims[0];
      for (int k = 1; k < N_; ++k) if (sims[k] > best) { best = sims[k]; idx = k; }
    }
    sIdx = idx; idxArr[b] = idx;
  }
  __syncthreads();
  const int idx = sIdx;
  dt_out[(size_t)b * N_ + n]     = (n == idx) ? 0.f : delta_in[(size_t)b * N_ + n] + 1.f;
  filled_out[(size_t)b * N_ + n] = (n == idx || emp[n] == 0) ? 1.f : 0.f;
}

// ---------------- scatter-update slots / cum_feats ----------------
__global__ void update_slots(const float* __restrict__ slots_in, const float* __restrict__ cum_in,
                             const float* __restrict__ x_t, const float* __restrict__ v,
                             const int* __restrict__ idxArr,
                             float* __restrict__ slots_out, float* __restrict__ cum_out) {
  const size_t i = (size_t)blockIdx.x * blockDim.x + threadIdx.x;
  if (i >= (size_t)B_ * N_ * D_) return;
  const int d = (int)(i & (D_ - 1));
  const int n = (int)((i >> 8) & (N_ - 1));
  const int b = (int)(i >> 14);
  const bool sel = (n == idxArr[b]);
  const float xv = x_t[(size_t)b * D_ + d];
  slots_out[i] = sel ? v[(size_t)b * D_ + d] : slots_in[i];
  cum_out[i]   = sel ? xv : cum_in[i] + xv;
}

// ---------------- pointwise LSTM cells ----------------
__global__ void lstm_point(const float* __restrict__ gates, float* __restrict__ h,
                           float* __restrict__ c, float* __restrict__ hCopy) {
  const size_t i = (size_t)blockIdx.x * blockDim.x + threadIdx.x;
  if (i >= (size_t)B_ * H_) return;
  const size_t b = i / H_, u = i % H_;
  const float* g = gates + b * (size_t)G_;
  const float ig = sigf(g[u]);
  const float fg = sigf(g[u + H_]);
  const float gg = tanhf(g[u + 2 * H_]);
  const float og = sigf(g[u + 3 * H_]);
  const float cn = fg * c[i] + ig * gg;
  const float hn = og * tanhf(cn);
  c[i] = cn; h[i] = hn;
  if (hCopy) hCopy[i] = hn;
}

__global__ void outer_point(const float* __restrict__ gates, const float* __restrict__ c_in,
                            float* __restrict__ h_out, float* __restrict__ c_out) {
  const size_t i = (size_t)blockIdx.x * blockDim.x + threadIdx.x;
  if (i >= (size_t)B_ * H_) return;
  const size_t b = i / H_, u = i % H_;
  const float* g = gates + b * (size_t)G_;
  const float ig = sigf(g[u]);
  const float fg = sigf(g[u + H_]);
  const float gg = tanhf(g[u + 2 * H_]);
  const float og = sigf(g[u + 3 * H_]);
  const float cn = fg * c_in[i] + ig * gg;
  c_out[i] = cn;
  h_out[i] = og * tanhf(cn);
}

// ---------------- host orchestration ----------------
extern "C" void kernel_launch(void* const* d_in, const int* in_sizes, int n_in,
                              void* d_out, int out_size, void* d_ws, size_t ws_size,
                              hipStream_t stream) {
  const float* x_t     = (const float*)d_in[0];
  const float* h_lstm  = (const float*)d_in[1];
  const float* c_lstm  = (const float*)d_in[2];
  // d_in[3] h_mem_prev: unused by the reference
  const float* slots   = (const float*)d_in[4];
  const float* cum     = (const float*)d_in[5];
  const float* delta_t = (const float*)d_in[6];
  const unsigned char* filled = (const unsigned char*)d_in[7];
  const float* Wq   = (const float*)d_in[8];
  const float* Wk   = (const float*)d_in[9];
  const float* Wv   = (const float*)d_in[10];
  const float* bv   = (const float*)d_in[11];
  const float* lWih = (const float*)d_in[12];
  const float* lWhh = (const float*)d_in[13];
  const float* lbih = (const float*)d_in[14];
  const float* lbhh = (const float*)d_in[15];
  const float* W_ih = (const float*)d_in[16];
  const float* b_ih = (const float*)d_in[17];
  const float* W_hh = (const float*)d_in[18];

  float* out = (float*)d_out;
  const size_t BH  = (size_t)B_ * H_;
  const size_t BND = (size_t)B_ * N_ * D_;
  const size_t BN  = (size_t)B_ * N_;
  float* o_hnew   = out;
  float* o_cnew   = out + BH;
  float* o_hmem   = out + 2 * BH;
  float* o_slots  = out + 3 * BH;
  float* o_cum    = o_slots + BND;
  float* o_dt     = o_cum + BND;
  float* o_filled = o_dt + BN;

  char* ws = (char*)d_ws;
  size_t off = 0;
  auto alloc = [&](size_t bytes) {
    char* p = ws + off;
    off += (bytes + 255) & ~(size_t)255;
    return p;
  };
  unsigned short* Wcomb1 = (unsigned short*)alloc((size_t)G_ * 1024 * 2); // [Whh | Wih[:, :512]]
  unsigned short* Wcomb2 = (unsigned short*)alloc((size_t)G_ * 1280 * 2); // [W_ih | W_hh]
  unsigned short* Wvb    = (unsigned short*)alloc((size_t)D_ * D_ * 2);
  unsigned short* Pb     = (unsigned short*)alloc((size_t)D_ * D_ * 2);
  float* wlast  = (float*)alloc((size_t)G_ * 4);
  float* bias1  = (float*)alloc((size_t)G_ * 4);
  float* qk     = (float*)alloc((size_t)B_ * D_ * 4);
  float* vbuf   = (float*)alloc((size_t)B_ * D_ * 4);
  int*   idxArr = (int*)alloc((size_t)B_ * 4);
  float* hstate = (float*)alloc(BH * 4);
  float* cstate = (float*)alloc(BH * 4);
  float* gates  = (float*)alloc((size_t)B_ * G_ * 4);

  // ---- weight prep (bf16 conversions, concats) ----
  cat2_bf16<<<1024, 256, 0, stream>>>(Wcomb1, lWhh, H_, H_, lWih, FEAT_, 2 * D_, G_);
  cat2_bf16<<<1024, 256, 0, stream>>>(Wcomb2, W_ih, D_ + H_, D_ + H_, W_hh, H_, H_, G_);
  cat2_bf16<<<256, 256, 0, stream>>>(Wvb, Wv, D_, D_, (const float*)nullptr, 0, 0, D_);
  prep_misc<<<(G_ + 255) / 256, 256, 0, stream>>>(lWih, lbih, lbhh, wlast, bias1);
  compute_P_bf16<<<dim3(16, 16), 256, 0, stream>>>(Wk, Wq, Pb);

  // ---- qk = x_t @ P^T ; v = x_t @ Wv^T + bv ----
  Seg sx{ x_t, (unsigned long)D_, 0, D_ };
  wmma_gemm_nt<<<dim3(D_ / 128, B_ / 128), 256, 0, stream>>>(
      sx, sx, sx, Pb, qk, D_, (const float*)nullptr, (const float*)nullptr,
      (const float*)nullptr, 0, D_, D_);
  wmma_gemm_nt<<<dim3(D_ / 128, B_ / 128), 256, 0, stream>>>(
      sx, sx, sx, Wvb, vbuf, D_, bv, (const float*)nullptr,
      (const float*)nullptr, 0, D_, D_);

  // ---- selection + scatter update ----
  select_slots<<<B_, 64, 0, stream>>>(qk, slots, delta_t, filled, o_dt, o_filled, idxArr);
  update_slots<<<(unsigned)(BND / 256), 256, 0, stream>>>(slots, cum, x_t, vbuf, idxArr,
                                                          o_slots, o_cum);

  // ---- slot LSTM over N=64 steps: fused GEMM over K = [h(512)|slots_t(256)|cum_t(256)] ----
  (void)hipMemsetAsync(hstate, 0, BH * 4, stream);
  (void)hipMemsetAsync(cstate, 0, BH * 4, stream);
  for (int t = 0; t < N_; ++t) {
    Seg s0{ hstate, (unsigned long)H_, 0, H_ };
    Seg s1{ o_slots + (size_t)t * D_, (unsigned long)N_ * D_, H_, H_ + D_ };
    Seg s2{ o_cum   + (size_t)t * D_, (unsigned long)N_ * D_, H_ + D_, H_ + 2 * D_ };
    wmma_gemm_nt<<<dim3(G_ / 128, B_ / 128), 256, 0, stream>>>(
        s0, s1, s2, Wcomb1, gates, G_, bias1, wlast, o_dt + t, (long)N_, 1024, G_);
    lstm_point<<<(unsigned)(BH / 256), 256, 0, stream>>>(
        gates, hstate, cstate, (t == N_ - 1) ? o_hmem : (float*)nullptr);
  }

  // ---- outer LSTM: K = [x_t(256)|h_mem(512)|h_lstm(512)] ----
  Seg t0{ x_t, (unsigned long)D_, 0, D_ };
  Seg t1{ o_hmem, (unsigned long)H_, D_, D_ + H_ };
  Seg t2{ h_lstm, (unsigned long)H_, D_ + H_, D_ + 2 * H_ };
  wmma_gemm_nt<<<dim3(G_ / 128, B_ / 128), 256, 0, stream>>>(
      t0, t1, t2, Wcomb2, gates, G_, b_ih, (const float*)nullptr,
      (const float*)nullptr, 0, 1280, G_);
  outer_point<<<(unsigned)(BH / 256), 256, 0, stream>>>(gates, c_lstm, o_hnew, o_cnew);
}
